// CausalAttention_9586367005246
// MI455X (gfx1250) — matrix operand order repaired and checked by reading, not compiled, not run on hardware
//
#include <hip/hip_runtime.h>
#include <hip/hip_bf16.h>
#include <stdint.h>

typedef __attribute__((ext_vector_type(16))) __bf16 v16bf;
typedef __attribute__((ext_vector_type(8)))  float  v8f;

#define DIN  512
#define DOUT 64

static __device__ __forceinline__ __bf16 tobf(float f) { return (__bf16)f; }

#if __has_builtin(__builtin_amdgcn_global_load_async_to_lds_b128) && __has_builtin(__builtin_amdgcn_s_wait_asynccnt)
#define USE_ASYNC_LDS 1
#endif

// 16-byte global->LDS copy: async (TDM-path ASYNCcnt-tracked) when available.
static __device__ __forceinline__ void cp16(const void* g, void* l) {
#ifdef USE_ASYNC_LDS
    typedef __attribute__((__vector_size__(16))) int i4;
    typedef __attribute__((address_space(1))) i4* g4p;
    typedef __attribute__((address_space(3))) i4* l4p;
    __builtin_amdgcn_global_load_async_to_lds_b128(
        (g4p)(uintptr_t)g, (l4p)(uint32_t)(uintptr_t)l, 0, 0);
#else
    *(uint4*)l = *(const uint4*)g;
#endif
}
static __device__ __forceinline__ void async_wait0() {
#ifdef USE_ASYNC_LDS
    __builtin_amdgcn_s_wait_asynccnt(0);
#endif
}

// ---------------------------------------------------------------------------
// QKV projection: q = 0.125 * x@Wq (bf16), k = x@Wk (bf16, row-major),
// vT = (x@Wv)^T (bf16, [B][DOUT][N]) so attention V-fragments are contiguous.
// 8 waves/block, one 16-row tile per wave; W staged in LDS pre-swizzled into
// B-fragment order (one contiguous 32B ds_load per lane per fragment).
// ---------------------------------------------------------------------------
__global__ __launch_bounds__(256) void qkv_proj_kernel(
    const float* __restrict__ x,
    const float* __restrict__ Wq, const float* __restrict__ Wk, const float* __restrict__ Wv,
    __bf16* __restrict__ qb, __bf16* __restrict__ kb, __bf16* __restrict__ vtb,
    int N)
{
    extern __shared__ __bf16 wlds[];   // [3][16 kk][4 nt][32 lane][16 e]
    const int tid = threadIdx.x;
    for (int i = tid; i < 3 * DIN * DOUT; i += 256) {
        const int w  = i / (DIN * DOUT);
        const int r  = i - w * (DIN * DOUT);
        const int k  = r >> 6;
        const int n  = r & 63;
        const int kk = k >> 5;
        const int km = k & 31;
        const int half = km >> 4;
        const int e    = km & 15;
        const int nt   = n >> 4;
        const int lane = (n & 15) + 16 * half;
        const float* Wp = (w == 0) ? Wq : ((w == 1) ? Wk : Wv);
        wlds[(((w * 16 + kk) * 4 + nt) * 32 + lane) * 16 + e] = tobf(Wp[k * DOUT + n]);
    }
    __syncthreads();

    const int wave = tid >> 5, lane = tid & 31;
    const int lhalf = lane >> 4, lanelo = lane & 15;
    const long rowbase = ((long)blockIdx.x * 8 + wave) * 16;
    const float* xrow = x + (rowbase + lanelo) * DIN;
    const long btile = rowbase / N;            // whole 16-row tile is one batch
    const long rn0   = rowbase - btile * N;

    v8f acc[3][4] = {};
    for (int kk = 0; kk < 16; ++kk) {
        v16bf a;
        const int d0 = kk * 32 + 16 * lhalf;
#pragma unroll
        for (int e = 0; e < 8; ++e) { a[e] = tobf(xrow[d0 + e]); a[8 + e] = tobf(xrow[d0 + 8 + e]); }
#pragma unroll
        for (int w = 0; w < 3; ++w) {
#pragma unroll
            for (int nt = 0; nt < 4; ++nt) {
                const v16bf b = *(const v16bf*)&wlds[(((w * 16 + kk) * 4 + nt) * 32 + lane) * 16];
                acc[w][nt] = __builtin_amdgcn_wmma_f32_16x16x32_bf16(
                    false, a, false, b, (short)0, acc[w][nt], false, false);
            }
        }
    }
    // q (pre-scaled) and k: row-major [B*N][DOUT]
#pragma unroll
    for (int w = 0; w < 2; ++w) {
        __bf16* op = (w == 0) ? qb : kb;
        const float sc = (w == 0) ? 0.125f : 1.0f;
#pragma unroll
        for (int nt = 0; nt < 4; ++nt)
#pragma unroll
            for (int r = 0; r < 8; ++r)
                op[(rowbase + r + 8 * lhalf) * DOUT + nt * 16 + lanelo] = tobf(acc[w][nt][r] * sc);
    }
    // v transposed: [B][DOUT][N]
#pragma unroll
    for (int nt = 0; nt < 4; ++nt)
#pragma unroll
        for (int r = 0; r < 8; ++r)
            vtb[(btile * DOUT + nt * 16 + lanelo) * N + rn0 + r + 8 * lhalf] = tobf(acc[2][nt][r]);
}

// ---------------------------------------------------------------------------
// Flash attention: 4 waves/block (64 q rows), 64-key tiles, double-buffered
// async K/vT staging so the async engine runs under WMMA + softmax work.
// ---------------------------------------------------------------------------
__global__ __launch_bounds__(128) void flash_attn_kernel(
    const __bf16* __restrict__ qb, const __bf16* __restrict__ kb, const __bf16* __restrict__ vtb,
    float* __restrict__ out, int N)
{
    __shared__ __attribute__((aligned(32))) __bf16 kt[2][64 * DOUT];    // [key][d]
    __shared__ __attribute__((aligned(32))) __bf16 vtt[2][DOUT * 64];   // [d][key]
    __shared__ __attribute__((aligned(32))) __bf16 pst[4][16 * 64];     // per-wave P

    const int tid = threadIdx.x;
    const int wave = tid >> 5, lane = tid & 31;
    const int lhalf = lane >> 4, lanelo = lane & 15;
    const int nqb  = N / 64;
    const int bidx = blockIdx.x / nqb;
    const int q0   = (blockIdx.x % nqb) * 64;
    const int qtile = q0 + wave * 16;

    const __bf16* qbase  = qb  + (long)bidx * N * DOUT;
    const __bf16* kbase  = kb  + (long)bidx * N * DOUT;
    const __bf16* vtbase = vtb + (long)bidx * DOUT * N;

    // q A-fragments (two 32-wide K chunks over Dout)
    v16bf aq[2];
    {
        const __bf16* qr = qbase + (long)(qtile + lanelo) * DOUT;
#pragma unroll
        for (int c = 0; c < 2; ++c) {
            const int d0 = c * 32 + 16 * lhalf;
#pragma unroll
            for (int e = 0; e < 8; ++e) { aq[c][e] = qr[d0 + e]; aq[c][8 + e] = qr[d0 + 8 + e]; }
        }
    }

    v8f acco[4] = {};
    float m[8], l[8];
#pragma unroll
    for (int r = 0; r < 8; ++r) { m[r] = -__builtin_inff(); l[r] = 0.0f; }

    __bf16* pw = pst[wave];

    // stage a 64-key tile (K row-major 8KB, vT 8KB) into buffer `buf`
    auto stage = [&](int buf, int j) {
        const uint4* ks = (const uint4*)(kbase + (long)j * DOUT);
        uint4* kd = (uint4*)kt[buf];
#pragma unroll
        for (int i = 0; i < 4; ++i) cp16(ks + tid + i * 128, kd + tid + i * 128);
#pragma unroll
        for (int i = 0; i < 4; ++i) {
            const int idx = tid + i * 128;
            const int d = idx >> 3, s = idx & 7;           // 8 x 16B per d-row
            cp16((const uint4*)(vtbase + (long)d * N + j) + s,
                 (uint4*)(vtt[buf] + d * 64) + s);
        }
    };

    const int nsteps = q0 / 64 + 1;
    stage(0, 0);
    async_wait0();
    __syncthreads();

    for (int s = 0; s < nsteps; ++s) {
        const int j = s * 64;
        const int cur = s & 1;
        if (s + 1 < nsteps) stage(1 - cur, j + 64);        // prefetch next tile

        if (j <= qtile + 15) {
            const __bf16* ktb = kt[cur];
            const __bf16* vtbL = vtt[cur];
            // ---- S = q @ k^T over 64 keys (4 groups of 16) ----
            v8f sg[4] = {};
#pragma unroll
            for (int c = 0; c < 2; ++c) {
                const int d0 = c * 32 + 16 * lhalf;
#pragma unroll
                for (int g = 0; g < 4; ++g) {
                    const v16bf bk = *(const v16bf*)(ktb + (g * 16 + lanelo) * DOUT + d0);
                    sg[g] = __builtin_amdgcn_wmma_f32_16x16x32_bf16(
                        false, aq[c], false, bk, (short)0, sg[g], false, false);
                }
            }
            // ---- online softmax (rowwise across 16 lanes) ----
#pragma unroll
            for (int r = 0; r < 8; ++r) {
                const int qrow = qtile + r + 8 * lhalf;
                float p[4];
                float mx = -__builtin_inff();
#pragma unroll
                for (int g = 0; g < 4; ++g) {
                    p[g] = (j + g * 16 + lanelo <= qrow) ? sg[g][r] : -__builtin_inff();
                    mx = fmaxf(mx, p[g]);
                }
#pragma unroll
                for (int off = 1; off < 16; off <<= 1) mx = fmaxf(mx, __shfl_xor(mx, off, 32));
                const float mnew = fmaxf(m[r], mx);
                float rs = 0.0f;
#pragma unroll
                for (int g = 0; g < 4; ++g) { p[g] = __expf(p[g] - mnew); rs += p[g]; }
#pragma unroll
                for (int off = 1; off < 16; off <<= 1) rs += __shfl_xor(rs, off, 32);
                const float al = __expf(m[r] - mnew);
                m[r] = mnew;
                l[r] = l[r] * al + rs;
#pragma unroll
                for (int nt = 0; nt < 4; ++nt) acco[nt][r] *= al;
#pragma unroll
                for (int g = 0; g < 4; ++g)
                    pw[(r + 8 * lhalf) * 64 + g * 16 + lanelo] = tobf(p[g]);
            }
            asm volatile("s_wait_dscnt 0" ::: "memory");   // same-wave DS RAW fence
            // ---- O += P @ V ----
            v16bf ap[2];
            {
                const __bf16* pr = pw + lanelo * 64;
#pragma unroll
                for (int c = 0; c < 2; ++c)
#pragma unroll
                    for (int e = 0; e < 8; ++e) {
                        ap[c][e]     = pr[c * 32 + 16 * lhalf + e];
                        ap[c][8 + e] = pr[c * 32 + 16 * lhalf + 8 + e];
                    }
            }
#pragma unroll
            for (int c = 0; c < 2; ++c)
#pragma unroll
                for (int nt = 0; nt < 4; ++nt) {
                    const v16bf bv = *(const v16bf*)(vtbL + (nt * 16 + lanelo) * 64 + c * 32 + 16 * lhalf);
                    acco[nt] = __builtin_amdgcn_wmma_f32_16x16x32_bf16(
                        false, ap[c], false, bv, (short)0, acco[nt], false, false);
                }
        }
        async_wait0();     // next tile landed during compute
        __syncthreads();
    }

    // normalize and write f32 output
    float* ob = out + ((long)bidx * N + qtile) * DOUT;
#pragma unroll
    for (int r = 0; r < 8; ++r) {
        const float inv = 1.0f / l[r];
        const int orow = r + 8 * lhalf;
#pragma unroll
        for (int nt = 0; nt < 4; ++nt)
            ob[orow * DOUT + nt * 16 + lanelo] = acco[nt][r] * inv;
    }
}

// ---------------------------------------------------------------------------
extern "C" void kernel_launch(void* const* d_in, const int* in_sizes, int n_in,
                              void* d_out, int out_size, void* d_ws, size_t ws_size,
                              hipStream_t stream) {
    const float* x  = (const float*)d_in[0];
    const float* Wq = (const float*)d_in[1];
    const float* Wk = (const float*)d_in[2];
    const float* Wv = (const float*)d_in[3];
    const long BN = (long)in_sizes[0] / DIN;   // B*N total rows
    const int  N  = 4096;

    __bf16* qb  = (__bf16*)d_ws;               // 0.125-scaled q
    __bf16* kb  = qb + BN * DOUT;
    __bf16* vtb = kb + BN * DOUT;              // v transposed [B][DOUT][N]

    const int tiles = (int)(BN / 16);
    qkv_proj_kernel<<<tiles / 8, 256, 3 * DIN * DOUT * sizeof(__bf16), stream>>>(
        x, Wq, Wk, Wv, qb, kb, vtb, N);
    flash_attn_kernel<<<(int)(BN / 64), 128, 0, stream>>>(qb, kb, vtb, (float*)d_out, N);
}
